// ProteinFeatures_32573031973440
// MI455X (gfx1250) — matrix operand order repaired and checked by reading, not compiled, not run on hardware
//
#include <hip/hip_runtime.h>
#include <hip/hip_bf16.h>

typedef __attribute__((ext_vector_type(16))) _Float16 v16h;
typedef __attribute__((ext_vector_type(8)))  _Float16 h8;
typedef __attribute__((ext_vector_type(8)))  float    v8f;

#define NB    2048
#define NROWS 8192          // B*N
#define TOPK  48
#define EIN   416
#define EF    128
#define FSTR  424           // padded LDS row stride (halves), 848B = 53*16B

// atom order: n=0, ca=1, c=2, o=3, cb=4
__constant__ int PA[24] = {0,2,3,4, 1,1,1,1, 0,0,0,4,4, 3,0,2,3,4, 2,3,4,2,3, 2};
__constant__ int PB[24] = {0,2,3,4, 0,2,3,4, 2,3,4,2,3, 2,1,1,1,1, 0,0,0,4,4, 3};

// ---------------- kernel 1: backbone atoms + virtual CB -> [row,15] ----------
__global__ __launch_bounds__(256) void atoms_kernel(const float* __restrict__ x,
                                                    float* __restrict__ atoms) {
  int i = blockIdx.x * 256 + threadIdx.x;
  if (i >= NROWS) return;
  const float* xi = x + (size_t)i * 12;
  float nx = xi[0], ny = xi[1], nz = xi[2];
  float ax = xi[3], ay = xi[4], az = xi[5];      // CA
  float cx = xi[6], cy = xi[7], cz = xi[8];
  float ox = xi[9], oy = xi[10], oz = xi[11];
  float bx = ax - nx, by = ay - ny, bz = az - nz;
  float gx = cx - ax, gy = cy - ay, gz = cz - az;
  float rx = by * gz - bz * gy;
  float ry = bz * gx - bx * gz;
  float rz = bx * gy - by * gx;
  float cbx = -0.58273431f * rx + 0.56802827f * bx - 0.54067466f * gx + ax;
  float cby = -0.58273431f * ry + 0.56802827f * by - 0.54067466f * gy + ay;
  float cbz = -0.58273431f * rz + 0.56802827f * bz - 0.54067466f * gz + az;
  float* o = atoms + (size_t)i * 15;
  o[0] = nx;  o[1] = ny;  o[2] = nz;
  o[3] = ax;  o[4] = ay;  o[5] = az;
  o[6] = cx;  o[7] = cy;  o[8] = cz;
  o[9] = ox;  o[10] = oy; o[11] = oz;
  o[12] = cbx; o[13] = cby; o[14] = cbz;
}

// ---------------- kernel 2: masked distances + top-48 (val asc, idx asc) -----
__global__ __launch_bounds__(256) void topk_kernel(const float* __restrict__ atoms,
                                                   const float* __restrict__ mask,
                                                   float* __restrict__ dnb,
                                                   int* __restrict__ eidx) {
  __shared__ float sd[NB];
  __shared__ float sm2[NB];
  __shared__ float sredf[256];
  __shared__ unsigned long long sred[256];
  const int row = blockIdx.x;
  const int b   = row >> 11;
  const int tid = threadIdx.x;
  const float cax = atoms[(size_t)row * 15 + 3];
  const float cay = atoms[(size_t)row * 15 + 4];
  const float caz = atoms[(size_t)row * 15 + 5];
  const float mi  = mask[row];
  float lmax = 0.0f;
  for (int j = tid; j < NB; j += 256) {
    int nj = (b << 11) + j;
    const float* aj = atoms + (size_t)nj * 15;
    float dx = cax - aj[3], dy = cay - aj[4], dz = caz - aj[5];
    float m2 = mi * mask[nj];
    float dd = m2 * sqrtf(dx * dx + dy * dy + dz * dz + 1e-6f);
    sd[j] = dd; sm2[j] = m2;
    lmax = fmaxf(lmax, dd);
  }
  sredf[tid] = lmax; __syncthreads();
  for (int s = 128; s > 0; s >>= 1) {
    if (tid < s) sredf[tid] = fmaxf(sredf[tid], sredf[tid + s]);
    __syncthreads();
  }
  const float dmax = sredf[0];
  for (int j = tid; j < NB; j += 256) sd[j] += (1.0f - sm2[j]) * dmax;
  __syncthreads();
  for (int k = 0; k < TOPK; ++k) {
    unsigned long long best = ~0ull;
    for (int j = tid; j < NB; j += 256) {
      // non-negative floats: bit pattern is order-preserving
      unsigned long long key =
          ((unsigned long long)__float_as_uint(sd[j]) << 32) | (unsigned)j;
      best = key < best ? key : best;
    }
    sred[tid] = best; __syncthreads();
    for (int s = 128; s > 0; s >>= 1) {
      if (tid < s) { unsigned long long o = sred[tid + s]; if (o < sred[tid]) sred[tid] = o; }
      __syncthreads();
    }
    unsigned long long win = sred[0];
    int jstar = (int)(win & 0xffffffffu);
    if (tid == 0) {
      eidx[(size_t)row * TOPK + k] = jstar;
      dnb [(size_t)row * TOPK + k] = __uint_as_float((unsigned)(win >> 32));
      sd[jstar] = 3.4e38f;
    }
    __syncthreads();
  }
}

// ---------------- kernel 3: w_edge f32 [416,128] -> f16 transposed [128,416] -
__global__ __launch_bounds__(256) void wt_kernel(const float* __restrict__ w_edge,
                                                 _Float16* __restrict__ wt) {
  int t = blockIdx.x * 256 + threadIdx.x;      // t = k*128 + n, t < 53248
  int n = t & 127, k = t >> 7;
  wt[(size_t)n * EIN + k] = (_Float16)w_edge[t];
}

// ---------------- kernel 4: fused features + WMMA GEMM + LayerNorm ----------
__global__ __launch_bounds__(256) void edge_kernel(
    const float* __restrict__ atoms, const float* __restrict__ dnb,
    const int* __restrict__ eidx, const int* __restrict__ ridx,
    const int* __restrict__ chain, const float* __restrict__ w_pos,
    const float* __restrict__ b_pos, const _Float16* __restrict__ wt,
    const float* __restrict__ gamma_, const float* __restrict__ beta_,
    float* __restrict__ eout) {
  __shared__ __align__(16) char smem_raw[TOPK * FSTR * 2];  // F(f16) then E(f32) alias
  _Float16* F = (_Float16*)smem_raw;
  float*    E = (float*)smem_raw;
  __shared__ float s_nb[TOPK * 15];
  __shared__ float s_ctr[15];
  __shared__ float s_dn[TOPK];
  __shared__ int   s_enc[TOPK];
  __shared__ int   s_ei[TOPK];
  __shared__ float s_mu[TOPK], s_rs[TOPK];

  const int row = blockIdx.x;      // b*N + i
  const int b   = row >> 11;
  const int tid = threadIdx.x;

  if (tid < 15) s_ctr[tid] = atoms[(size_t)row * 15 + tid];
  if (tid < TOPK) {
    int j  = eidx[(size_t)row * TOPK + tid];
    s_ei[tid] = j;
    s_dn[tid] = dnb[(size_t)row * TOPK + tid];
    int nj   = (b << 11) + j;
    int off  = ridx[row] - ridx[nj];
    int same = (chain[row] == chain[nj]) ? 1 : 0;
    int v = off + 32; v = v < 0 ? 0 : (v > 64 ? 64 : v);
    s_enc[tid] = same ? v : 65;
  }
  __syncthreads();
  for (int t = tid; t < TOPK * 15; t += 256) {
    int k = t / 15, c = t - k * 15;
    s_nb[t] = atoms[((size_t)((b << 11) + s_ei[k])) * 15 + c];
  }
  __syncthreads();

  // ---- feature tile F[48][416] (f16), 26 groups of 16 columns per edge ----
  for (int t = tid; t < TOPK * 26; t += 256) {
    int k = t / 26, g = t - k * 26;
    _Float16* frow = F + k * FSTR + g * 16;
    if (g == 0) {                                    // positional embedding
      const float* wr = w_pos + (size_t)s_enc[k] * 16;
      #pragma unroll
      for (int j = 0; j < 16; ++j) frow[j] = (_Float16)(wr[j] + b_pos[j]);
    } else {
      float d;
      if (g == 1) {                                  // CA-CA uses topk distance
        d = s_dn[k];
      } else {
        int p = g - 2;
        int an = PA[p] * 3, bn = PB[p] * 3;
        float dx = s_ctr[an]     - s_nb[k * 15 + bn];
        float dy = s_ctr[an + 1] - s_nb[k * 15 + bn + 1];
        float dz = s_ctr[an + 2] - s_nb[k * 15 + bn + 2];
        d = sqrtf(dx * dx + dy * dy + dz * dz + 1e-6f);
      }
      #pragma unroll
      for (int j = 0; j < 16; ++j) {                 // rbf: mu in [2,22], sigma=1.25
        float u = (d - (2.0f + 1.33333333f * (float)j)) * 0.8f;
        frow[j] = (_Float16)__expf(-u * u);
      }
    }
  }
  __syncthreads();

  // ---- GEMM: [48 x 416] @ [416 x 128], f16 in / f32 acc, WMMA 16x16x32 ----
  const int wave = tid >> 5;
  const int lane = tid & 31;
  const int ln   = lane & 15;
  const int half = lane >> 4;
  const int ncol = wave * 16 + ln;
  const _Float16* wbase = wt + (size_t)ncol * EIN;

  v8f c[3]; c[0] = (v8f)0.0f; c[1] = (v8f)0.0f; c[2] = (v8f)0.0f;

  #pragma unroll
  for (int kk = 0; kk < 13; ++kk) {
    const int k0 = kk * 32;
    // B fragment: col=ncol, K = k0 + 16*half + e (e=0..15), contiguous in wt
    h8 blo = *(const h8*)(wbase + k0 + 16 * half);
    h8 bhi = *(const h8*)(wbase + k0 + 16 * half + 8);
    v16h bf = __builtin_shufflevector(blo, bhi, 0,1,2,3,4,5,6,7,8,9,10,11,12,13,14,15);
    #pragma unroll
    for (int mt = 0; mt < 3; ++mt) {
      // A fragment: row = mt*16+ln, K = k0 + 16*half + e (e=0..15)
      const _Float16* ab = F + (mt * 16 + ln) * FSTR + k0 + 16 * half;
      h8 alo = *(const h8*)ab;
      h8 ahi = *(const h8*)(ab + 8);
      v16h af = __builtin_shufflevector(alo, ahi, 0,1,2,3,4,5,6,7,8,9,10,11,12,13,14,15);
      c[mt] = __builtin_amdgcn_wmma_f32_16x16x32_f16(
          false, af, false, bf, (short)0, c[mt], false, false);
    }
  }
  __syncthreads();   // all waves done reading F; safe to alias with E

  #pragma unroll
  for (int mt = 0; mt < 3; ++mt)
    #pragma unroll
    for (int v = 0; v < 8; ++v)
      E[(mt * 16 + half * 8 + v) * EF + ncol] = c[mt][v];  // D: M = v + 8*half
  __syncthreads();

  // ---- LayerNorm over 128 features per edge ----
  if (tid < TOPK) {
    const float* er = E + tid * EF;
    float s = 0.0f, ss = 0.0f;
    for (int j = 0; j < EF; ++j) { float x = er[j]; s += x; ss += x * x; }
    float mu  = s * (1.0f / EF);
    float var = ss * (1.0f / EF) - mu * mu;
    s_mu[tid] = mu;
    s_rs[tid] = rsqrtf(var + 1e-5f);
  }
  __syncthreads();
  float* orow = eout + (size_t)row * (TOPK * EF);
  for (int t = tid; t < TOPK * EF; t += 256) {
    int r = t >> 7, j = t & 127;
    orow[t] = (E[t] - s_mu[r]) * s_rs[r] * gamma_[j] + beta_[j];
  }
}

extern "C" void kernel_launch(void* const* d_in, const int* in_sizes, int n_in,
                              void* d_out, int out_size, void* d_ws, size_t ws_size,
                              hipStream_t stream) {
  const float* x      = (const float*)d_in[0];
  const float* mask   = (const float*)d_in[1];
  const int*   ridx   = (const int*)  d_in[2];
  const int*   chain  = (const int*)  d_in[3];
  const float* w_pos  = (const float*)d_in[4];
  const float* b_pos  = (const float*)d_in[5];
  const float* w_edge = (const float*)d_in[6];
  const float* gam    = (const float*)d_in[7];
  const float* bet    = (const float*)d_in[8];

  // workspace layout
  float* atoms = (float*)d_ws;                           // 8192*15 f32
  float* dnb   = atoms + (size_t)NROWS * 15;             // 8192*48 f32
  _Float16* wt = (_Float16*)(dnb + (size_t)NROWS * TOPK); // 128*416 f16 (16B aligned)

  // outputs: e [8192*48*128] f32, then e_idx [8192*48] i32
  float* eout   = (float*)d_out;
  int*   ei_out = (int*)d_out + (size_t)NROWS * TOPK * EF;

  atoms_kernel<<<NROWS / 256, 256, 0, stream>>>(x, atoms);
  wt_kernel<<<(EIN * EF) / 256, 256, 0, stream>>>(w_edge, wt);
  topk_kernel<<<NROWS, 256, 0, stream>>>(atoms, mask, dnb, ei_out);
  edge_kernel<<<NROWS, 256, 0, stream>>>(atoms, dnb, ei_out, ridx, chain,
                                         w_pos, b_pos, wt, gam, bet, eout);
}
